// FraudGNN_42588895707795
// MI455X (gfx1250) — matrix operand order repaired and checked by reading, not compiled, not run on hardware
//
#include <hip/hip_runtime.h>

// ---------------------------------------------------------------------------
// FraudGNN for MI455X (gfx1250, wave32).
// GEMMs: v_wmma_f32_16x16x32_bf16 (f32 accumulate), software-pipelined K-loop,
// dual-accumulate (SAGE: agg@Wl + x@Wr) as a compile-time template so no
// EXEC-masked branch sits between the two WMMAs. Sparse aggregation: global
// f32 atomics into L2-resident buffers (agg 51MB / gat_out 102MB fit the
// 192MB L2). Edge softmax via ordered-uint atomicMax.
// ---------------------------------------------------------------------------

#define NNODES 50000
#define NEDGES 600000
#define INC    256
#define HIDC   128
#define NHEADS 4

typedef __attribute__((ext_vector_type(16))) __bf16 v16bf;
typedef __attribute__((ext_vector_type(8)))  __bf16 v8bf;
typedef __attribute__((ext_vector_type(8)))  float  v8f;

__device__ __forceinline__ __bf16 f2bf(float f) {
    unsigned u = __builtin_bit_cast(unsigned, f);
    u += 0x7FFFu + ((u >> 16) & 1u);              // round-to-nearest-even
    return __builtin_bit_cast(__bf16, (unsigned short)(u >> 16));
}
// monotone float<->uint encoding for atomicMax-based segment max
__device__ __forceinline__ unsigned enc_f(float f) {
    unsigned u = __builtin_bit_cast(unsigned, f);
    return (u & 0x80000000u) ? ~u : (u | 0x80000000u);
}
__device__ __forceinline__ float dec_f(unsigned u) {
    unsigned v = (u & 0x80000000u) ? (u & 0x7FFFFFFFu) : ~u;
    return __builtin_bit_cast(float, v);
}

__device__ __forceinline__ v16bf pack16(v8bf lo, v8bf hi) {
    v16bf r;
#pragma unroll
    for (int i = 0; i < 8; ++i) { r[i] = lo[i]; r[i + 8] = hi[i]; }
    return r;
}

// ---------------------------------------------------------------------------
// WMMA GEMM:  C[M,N] = A1[M,K] @ B1t[N,K]^T  (+ A2 @ B2t^T) (+ bias) (+ relu)
// One wave per 16x16 output tile. Fragment layouts per CDNA5 ISA 7.12.2.
// K-loop software-pipelined: next step's fragments are issued before the
// WMMAs consuming the current fragments, so loads overlap the matrix pipe.
// ---------------------------------------------------------------------------
template <bool DUAL>
__global__ __launch_bounds__(256)
void gemm_wmma_bf16(const __bf16* __restrict__ A1, const __bf16* __restrict__ B1t,
                    const __bf16* __restrict__ A2, const __bf16* __restrict__ B2t,
                    const float* __restrict__ bias, float* __restrict__ C,
                    int M, int K, int Nout, int relu)
{
    const int wid  = (blockIdx.x * blockDim.x + threadIdx.x) >> 5;
    const int lane = threadIdx.x & 31;
    const int ntn  = Nout >> 4;
    const int tm   = wid / ntn;
    const int tn   = wid - tm * ntn;
    if (tm >= (M >> 4)) return;

    const int half = lane >> 4;
    const int r    = lane & 15;

    const __bf16* a1p = A1 + (size_t)(tm * 16 + r) * K + half * 8;
    const __bf16* b1p = B1t + (size_t)(tn * 16 + r) * K + half * 8;
    const __bf16* a2p = DUAL ? A2 + (size_t)(tm * 16 + r) * K + half * 8 : nullptr;
    const __bf16* b2p = DUAL ? B2t + (size_t)(tn * 16 + r) * K + half * 8 : nullptr;

    // prologue: fragments for k0 = 0
    v8bf alo = *(const v8bf*)(a1p);
    v8bf ahi = *(const v8bf*)(a1p + 16);
    v8bf blo = *(const v8bf*)(b1p);
    v8bf bhi = *(const v8bf*)(b1p + 16);
    v8bf alo2 = alo, ahi2 = ahi, blo2 = blo, bhi2 = bhi;
    if (DUAL) {
        alo2 = *(const v8bf*)(a2p);
        ahi2 = *(const v8bf*)(a2p + 16);
        blo2 = *(const v8bf*)(b2p);
        bhi2 = *(const v8bf*)(b2p + 16);
    }

    v8f acc = {};
    int k0 = 0;
    for (; k0 + 32 < K; k0 += 32) {
        const int kn = k0 + 32;
        __builtin_prefetch(a1p + kn + 32, 0, 1);   // global_prefetch_b8
        v8bf nalo = *(const v8bf*)(a1p + kn);
        v8bf nahi = *(const v8bf*)(a1p + kn + 16);
        v8bf nblo = *(const v8bf*)(b1p + kn);
        v8bf nbhi = *(const v8bf*)(b1p + kn + 16);
        v8bf nalo2 = nalo, nahi2 = nahi, nblo2 = nblo, nbhi2 = nbhi;
        if (DUAL) {
            nalo2 = *(const v8bf*)(a2p + kn);
            nahi2 = *(const v8bf*)(a2p + kn + 16);
            nblo2 = *(const v8bf*)(b2p + kn);
            nbhi2 = *(const v8bf*)(b2p + kn + 16);
        }
        acc = __builtin_amdgcn_wmma_f32_16x16x32_bf16(false, pack16(alo, ahi),
                                                      false, pack16(blo, bhi),
                                                      (short)0, acc, false, false);
        if (DUAL) {
            acc = __builtin_amdgcn_wmma_f32_16x16x32_bf16(false, pack16(alo2, ahi2),
                                                          false, pack16(blo2, bhi2),
                                                          (short)0, acc, false, false);
        }
        alo = nalo; ahi = nahi; blo = nblo; bhi = nbhi;
        if (DUAL) { alo2 = nalo2; ahi2 = nahi2; blo2 = nblo2; bhi2 = nbhi2; }
    }
    acc = __builtin_amdgcn_wmma_f32_16x16x32_bf16(false, pack16(alo, ahi),
                                                  false, pack16(blo, bhi),
                                                  (short)0, acc, false, false);
    if (DUAL) {
        acc = __builtin_amdgcn_wmma_f32_16x16x32_bf16(false, pack16(alo2, ahi2),
                                                      false, pack16(blo2, bhi2),
                                                      (short)0, acc, false, false);
    }

    const int   col = tn * 16 + r;
    const float bv  = bias ? bias[col] : 0.0f;
#pragma unroll
    for (int j = 0; j < 8; ++j) {
        int   row = tm * 16 + j + half * 8;
        float v   = acc[j] + bv;
        if (relu) v = fmaxf(v, 0.0f);
        C[(size_t)row * Nout + col] = v;
    }
}

// --------------------------- small helper kernels ---------------------------
__global__ void cvt_bf16_kernel(const float* __restrict__ X, __bf16* __restrict__ Y, int n) {
    int t = blockIdx.x * 256 + threadIdx.x;
    if (t < n) Y[t] = f2bf(X[t]);
}

__global__ void transpose_cvt_kernel(const float* __restrict__ W, __bf16* __restrict__ Wt,
                                     int K, int Nout) {
    int t = blockIdx.x * 256 + threadIdx.x;
    if (t >= K * Nout) return;
    int n = t / K, k = t - n * K;
    Wt[t] = f2bf(W[(size_t)k * Nout + n]);
}

__global__ void deg_kernel(const long long* __restrict__ dst, float* __restrict__ deg, int E) {
    int e = blockIdx.x * 256 + threadIdx.x;
    if (e < E) atomicAdd(&deg[(int)dst[e]], 1.0f);
}

__global__ void scatter_add_kernel(const float* __restrict__ X,
                                   const long long* __restrict__ src,
                                   const long long* __restrict__ dst,
                                   float* __restrict__ agg, int E, int C) {
    int t = blockIdx.x * 256 + threadIdx.x;
    int groups = C >> 2;
    int e = t / groups;
    if (e >= E) return;
    int g = (t - e * groups) << 2;
    int s = (int)src[e], d = (int)dst[e];
    const float* xp = X + (size_t)s * C + g;
    float*       ap = agg + (size_t)d * C + g;
#pragma unroll
    for (int i = 0; i < 4; ++i) atomicAdd(ap + i, xp[i]);
}

__global__ void div_cvt_kernel(const float* __restrict__ agg, const float* __restrict__ deg,
                               __bf16* __restrict__ out, int Nn, int C) {
    int t = blockIdx.x * 256 + threadIdx.x;
    if (t >= Nn * C) return;
    int row = t / C;
    out[t] = f2bf(agg[t] / fmaxf(deg[row], 1.0f));
}

__global__ void bn_stats_kernel(const float* __restrict__ X, int M, int C,
                                const float* __restrict__ g, const float* __restrict__ b,
                                float* __restrict__ scale, float* __restrict__ shift) {
    __shared__ float ssum[256], ssq[256];
    int c = blockIdx.x;
    float s = 0.f, q = 0.f;
    for (int i = threadIdx.x; i < M; i += 256) {
        float v = X[(size_t)i * C + c];
        s += v; q += v * v;
    }
    ssum[threadIdx.x] = s; ssq[threadIdx.x] = q;
    __syncthreads();
    for (int w = 128; w > 0; w >>= 1) {
        if (threadIdx.x < w) { ssum[threadIdx.x] += ssum[threadIdx.x + w];
                               ssq[threadIdx.x]  += ssq[threadIdx.x + w]; }
        __syncthreads();
    }
    if (threadIdx.x == 0) {
        float mu  = ssum[0] / (float)M;
        float var = ssq[0] / (float)M - mu * mu;
        float sc  = g[c] * rsqrtf(var + 1e-5f);
        scale[c] = sc;
        shift[c] = b[c] - mu * sc;
    }
}

__global__ void bn_apply_kernel(const float* __restrict__ X,
                                const float* __restrict__ scale, const float* __restrict__ shift,
                                float* __restrict__ Hf, __bf16* __restrict__ Hb,
                                int n, int C) {
    int t = blockIdx.x * 256 + threadIdx.x;
    if (t >= n) return;
    int c = t % C;
    float v = fmaxf(X[t] * scale[c] + shift[c], 0.0f);
    Hf[t] = v;
    Hb[t] = f2bf(v);
}

__global__ void att_scores_kernel(const float* __restrict__ xh,
                                  const float* __restrict__ att_s, const float* __restrict__ att_d,
                                  float* __restrict__ a_s, float* __restrict__ a_d, int Nn) {
    int t = blockIdx.x * 256 + threadIdx.x;
    if (t >= Nn * NHEADS) return;
    int n = t >> 2, h = t & 3;
    const float* xr = xh + (size_t)n * (NHEADS * HIDC) + h * HIDC;
    const float* as = att_s + h * HIDC;
    const float* ad = att_d + h * HIDC;
    float s = 0.f, d = 0.f;
    for (int c = 0; c < HIDC; ++c) { float v = xr[c]; s += v * as[c]; d += v * ad[c]; }
    a_s[t] = s; a_d[t] = d;
}

__global__ void gat_pass1_kernel(const long long* __restrict__ src, const long long* __restrict__ dst,
                                 const float* __restrict__ a_s, const float* __restrict__ a_d,
                                 float* __restrict__ e_buf, unsigned* __restrict__ m_u,
                                 int E, int Nn) {
    int t = blockIdx.x * 256 + threadIdx.x;
    if (t >= (E + Nn) * NHEADS) return;
    int ei = t >> 2, h = t & 3;
    int s = ei < E ? (int)src[ei] : ei - E;
    int d = ei < E ? (int)dst[ei] : ei - E;
    float ev = a_s[s * NHEADS + h] + a_d[d * NHEADS + h];
    ev = ev > 0.f ? ev : 0.2f * ev;
    e_buf[t] = ev;
    atomicMax(&m_u[d * NHEADS + h], enc_f(ev));
}

__global__ void gat_pass2_kernel(const long long* __restrict__ dst,
                                 float* __restrict__ e_buf, const unsigned* __restrict__ m_u,
                                 float* __restrict__ z, int E, int Nn) {
    int t = blockIdx.x * 256 + threadIdx.x;
    if (t >= (E + Nn) * NHEADS) return;
    int ei = t >> 2, h = t & 3;
    int d = ei < E ? (int)dst[ei] : ei - E;
    float w = __expf(e_buf[t] - dec_f(m_u[d * NHEADS + h]));
    e_buf[t] = w;
    atomicAdd(&z[d * NHEADS + h], w);
}

__global__ void gat_pass3_kernel(const long long* __restrict__ src, const long long* __restrict__ dst,
                                 const float* __restrict__ xh, const float* __restrict__ e_buf,
                                 const float* __restrict__ z, float* __restrict__ gout,
                                 int E, int Nn) {
    int t = blockIdx.x * 256 + threadIdx.x;
    if (t >= (E + Nn) * HIDC) return;
    int ei = t >> 7, c = t & (HIDC - 1);
    int s = ei < E ? (int)src[ei] : ei - E;
    int d = ei < E ? (int)dst[ei] : ei - E;
#pragma unroll
    for (int h = 0; h < NHEADS; ++h) {
        float alpha = e_buf[ei * NHEADS + h] / (z[d * NHEADS + h] + 1e-16f);
        atomicAdd(&gout[(size_t)d * (NHEADS * HIDC) + h * HIDC + c],
                  xh[(size_t)s * (NHEADS * HIDC) + h * HIDC + c] * alpha);
    }
}

__global__ void gat_mean_kernel(const float* __restrict__ gout, const float* __restrict__ bg,
                                float* __restrict__ pre, int Nn) {
    int t = blockIdx.x * 256 + threadIdx.x;
    if (t >= Nn * HIDC) return;
    int n = t >> 7, c = t & (HIDC - 1);
    const float* gp = gout + (size_t)n * (NHEADS * HIDC) + c;
    pre[t] = 0.25f * (gp[0] + gp[HIDC] + gp[2 * HIDC] + gp[3 * HIDC]) + bg[c];
}

__global__ void fc2_kernel(const float* __restrict__ fc1, const float* __restrict__ W,
                           const float* __restrict__ b, float* __restrict__ out, int Nn) {
    int n = blockIdx.x * 256 + threadIdx.x;
    if (n >= Nn) return;
    const float* f = fc1 + (size_t)n * 64;
    float a0 = b[0], a1 = b[1];
    for (int k = 0; k < 64; ++k) { float v = f[k]; a0 += v * W[2 * k]; a1 += v * W[2 * k + 1]; }
    out[n * 2]     = a0;
    out[n * 2 + 1] = a1;
}

// ---------------------------------------------------------------------------
static inline unsigned cdiv(size_t a, size_t b) { return (unsigned)((a + b - 1) / b); }

extern "C" void kernel_launch(void* const* d_in, const int* in_sizes, int n_in,
                              void* d_out, int out_size, void* d_ws, size_t ws_size,
                              hipStream_t stream) {
    const int Nn = NNODES, Ee = NEDGES;

    const float*     x    = (const float*)d_in[0];
    const long long* eidx = (const long long*)d_in[1];
    const long long* src  = eidx;
    const long long* dst  = eidx + Ee;
    const float* Wl1 = (const float*)d_in[2];  const float* bl1  = (const float*)d_in[3];
    const float* Wr1 = (const float*)d_in[4];
    const float* g1  = (const float*)d_in[5];  const float* b1   = (const float*)d_in[6];
    const float* Wl2 = (const float*)d_in[7];  const float* bl2  = (const float*)d_in[8];
    const float* Wr2 = (const float*)d_in[9];
    const float* g2  = (const float*)d_in[10]; const float* b2   = (const float*)d_in[11];
    const float* Wg  = (const float*)d_in[12];
    const float* atS = (const float*)d_in[13]; const float* atD  = (const float*)d_in[14];
    const float* bg  = (const float*)d_in[15];
    const float* gg  = (const float*)d_in[16]; const float* gb   = (const float*)d_in[17];
    const float* Wf1 = (const float*)d_in[18]; const float* bf1  = (const float*)d_in[19];
    const float* Wf2 = (const float*)d_in[20]; const float* bf2  = (const float*)d_in[21];

    char* ws = (char*)d_ws;
    size_t off = 0;
    auto bump = [&](size_t bytes) -> void* {
        void* p = ws + off;
        off += (bytes + 255) & ~(size_t)255;
        return p;
    };
    __bf16* x_bf = (__bf16*)bump((size_t)Nn * INC * 2);
    float*  aggf = (float*) bump((size_t)Nn * INC * 4);
    __bf16* aggb = (__bf16*)bump((size_t)Nn * INC * 2);
    float*  gat_out = (float*)x_bf;
    float*  fc1o    = (float*)x_bf;

    float*  xh  = (float*) bump((size_t)Nn * NHEADS * HIDC * 4);
    float*  pre = xh;
    float*  hf  = (float*) bump((size_t)Nn * HIDC * 4);
    __bf16* hb  = (__bf16*)bump((size_t)Nn * HIDC * 2);
    float*  deg = (float*) bump((size_t)Nn * 4);
    __bf16* Wl1t = (__bf16*)bump((size_t)HIDC * INC * 2);
    __bf16* Wr1t = (__bf16*)bump((size_t)HIDC * INC * 2);
    __bf16* Wl2t = (__bf16*)bump((size_t)HIDC * HIDC * 2);
    __bf16* Wr2t = (__bf16*)bump((size_t)HIDC * HIDC * 2);
    __bf16* Wgt  = (__bf16*)bump((size_t)NHEADS * HIDC * HIDC * 2);
    __bf16* Wf1t = (__bf16*)bump((size_t)64 * HIDC * 2);
    float* scale = (float*)bump(512 * 4);
    float* shift = (float*)bump(512 * 4);
    float* a_s   = (float*)bump((size_t)Nn * NHEADS * 4);
    float* a_d   = (float*)bump((size_t)Nn * NHEADS * 4);
    float* e_buf = (float*)bump((size_t)(Ee + Nn) * NHEADS * 4);
    unsigned* m_u = (unsigned*)bump((size_t)Nn * NHEADS * 4);
    float* z     = (float*)bump((size_t)Nn * NHEADS * 4);

    transpose_cvt_kernel<<<cdiv((size_t)INC * HIDC, 256), 256, 0, stream>>>(Wl1, Wl1t, INC, HIDC);
    transpose_cvt_kernel<<<cdiv((size_t)INC * HIDC, 256), 256, 0, stream>>>(Wr1, Wr1t, INC, HIDC);
    transpose_cvt_kernel<<<cdiv((size_t)HIDC * HIDC, 256), 256, 0, stream>>>(Wl2, Wl2t, HIDC, HIDC);
    transpose_cvt_kernel<<<cdiv((size_t)HIDC * HIDC, 256), 256, 0, stream>>>(Wr2, Wr2t, HIDC, HIDC);
    transpose_cvt_kernel<<<cdiv((size_t)HIDC * NHEADS * HIDC, 256), 256, 0, stream>>>(Wg, Wgt, HIDC, NHEADS * HIDC);
    transpose_cvt_kernel<<<cdiv((size_t)HIDC * 64, 256), 256, 0, stream>>>(Wf1, Wf1t, HIDC, 64);
    cvt_bf16_kernel<<<cdiv((size_t)Nn * INC, 256), 256, 0, stream>>>(x, x_bf, Nn * INC);

    (void)hipMemsetAsync(deg, 0, (size_t)Nn * 4, stream);
    deg_kernel<<<cdiv(Ee, 256), 256, 0, stream>>>(dst, deg, Ee);

    (void)hipMemsetAsync(aggf, 0, (size_t)Nn * INC * 4, stream);
    scatter_add_kernel<<<cdiv((size_t)Ee * (INC / 4), 256), 256, 0, stream>>>(x, src, dst, aggf, Ee, INC);
    div_cvt_kernel<<<cdiv((size_t)Nn * INC, 256), 256, 0, stream>>>(aggf, deg, aggb, Nn, INC);
    {   unsigned waves = (Nn / 16) * (HIDC / 16);
        gemm_wmma_bf16<true><<<cdiv(waves, 8), 256, 0, stream>>>(aggb, Wl1t, x_bf, Wr1t, bl1, pre, Nn, INC, HIDC, 0);
    }
    bn_stats_kernel<<<HIDC, 256, 0, stream>>>(pre, Nn, HIDC, g1, b1, scale, shift);
    bn_apply_kernel<<<cdiv((size_t)Nn * HIDC, 256), 256, 0, stream>>>(pre, scale, shift, hf, hb, Nn * HIDC, HIDC);

    (void)hipMemsetAsync(aggf, 0, (size_t)Nn * HIDC * 4, stream);
    scatter_add_kernel<<<cdiv((size_t)Ee * (HIDC / 4), 256), 256, 0, stream>>>(hf, src, dst, aggf, Ee, HIDC);
    div_cvt_kernel<<<cdiv((size_t)Nn * HIDC, 256), 256, 0, stream>>>(aggf, deg, aggb, Nn, HIDC);
    {   unsigned waves = (Nn / 16) * (HIDC / 16);
        gemm_wmma_bf16<true><<<cdiv(waves, 8), 256, 0, stream>>>(aggb, Wl2t, hb, Wr2t, bl2, pre, Nn, HIDC, HIDC, 0);
    }
    bn_stats_kernel<<<HIDC, 256, 0, stream>>>(pre, Nn, HIDC, g2, b2, scale, shift);
    bn_apply_kernel<<<cdiv((size_t)Nn * HIDC, 256), 256, 0, stream>>>(pre, scale, shift, hf, hb, Nn * HIDC, HIDC);

    {   unsigned waves = (Nn / 16) * ((NHEADS * HIDC) / 16);
        gemm_wmma_bf16<false><<<cdiv(waves, 8), 256, 0, stream>>>(hb, Wgt, nullptr, nullptr, nullptr, xh, Nn, HIDC, NHEADS * HIDC, 0);
    }
    att_scores_kernel<<<cdiv((size_t)Nn * NHEADS, 256), 256, 0, stream>>>(xh, atS, atD, a_s, a_d, Nn);
    (void)hipMemsetAsync(m_u, 0, (size_t)Nn * NHEADS * 4, stream);
    (void)hipMemsetAsync(z,   0, (size_t)Nn * NHEADS * 4, stream);
    gat_pass1_kernel<<<cdiv((size_t)(Ee + Nn) * NHEADS, 256), 256, 0, stream>>>(src, dst, a_s, a_d, e_buf, m_u, Ee, Nn);
    gat_pass2_kernel<<<cdiv((size_t)(Ee + Nn) * NHEADS, 256), 256, 0, stream>>>(dst, e_buf, m_u, z, Ee, Nn);
    (void)hipMemsetAsync(gat_out, 0, (size_t)Nn * NHEADS * HIDC * 4, stream);
    gat_pass3_kernel<<<cdiv((size_t)(Ee + Nn) * HIDC, 256), 256, 0, stream>>>(src, dst, xh, e_buf, z, gat_out, Ee, Nn);
    gat_mean_kernel<<<cdiv((size_t)Nn * HIDC, 256), 256, 0, stream>>>(gat_out, bg, pre, Nn);
    bn_stats_kernel<<<HIDC, 256, 0, stream>>>(pre, Nn, HIDC, gg, gb, scale, shift);
    bn_apply_kernel<<<cdiv((size_t)Nn * HIDC, 256), 256, 0, stream>>>(pre, scale, shift, hf, hb, Nn * HIDC, HIDC);

    {   unsigned waves = (Nn / 16) * (64 / 16);
        gemm_wmma_bf16<false><<<cdiv(waves, 8), 256, 0, stream>>>(hb, Wf1t, nullptr, nullptr, bf1, fc1o, Nn, HIDC, 64, 1);
    }
    fc2_kernel<<<cdiv(Nn, 256), 256, 0, stream>>>(fc1o, Wf2, bf2, (float*)d_out, Nn);
}
